// AttentionHead_16664473109085
// MI455X (gfx1250) — compile-verified, matrix operand order checked by reading, not run on hardware
//
#include <hip/hip_runtime.h>

// Problem constants (match reference)
constexpr int Bn = 4;
constexpr int Sn = 8192;
constexpr int Dn = 512;          // DIN == DQ == DK == 512
constexpr int NGL = 64;
constexpr int TILE_K = 32;       // WMMA K per step (bf16 16x16x32)
constexpr int LDSS = 40;         // padded LDS row stride (halfwords) for row-major tiles
constexpr int SPLITK = 8;        // deterministic split-K for the scores GEMM
constexpr int TRS = 130;         // TDM slab row stride in halfwords (256B row + 4B pad)

typedef __attribute__((ext_vector_type(16))) __bf16 v16bf;
typedef __attribute__((ext_vector_type(2)))  __bf16 v2bf;
typedef __attribute__((ext_vector_type(8)))  float  v8f;
typedef __attribute__((ext_vector_type(4)))  unsigned v4u;
typedef __attribute__((ext_vector_type(8)))  unsigned v8u;

// ---------- helpers ----------
__device__ __forceinline__ unsigned short f2bf(float f) {
    unsigned u = __float_as_uint(f);
    unsigned r = (u + 0x7FFFu + ((u >> 16) & 1u)) >> 16;   // round-to-nearest-even
    return (unsigned short)r;
}
__device__ __forceinline__ unsigned pack2(float a, float b) {
#if __has_builtin(__builtin_amdgcn_cvt_pk_bf16_f32)
    v2bf p = __builtin_amdgcn_cvt_pk_bf16_f32(a, b);       // v_cvt_pk_bf16_f32 (RNE)
    return __builtin_bit_cast(unsigned, p);
#else
    unsigned r;
    asm("v_cvt_pk_bf16_f32 %0, %1, %2" : "=v"(r) : "v"(a), "v"(b));
    return r;
#endif
}

// Async global -> LDS copy of one 16B chunk per lane (GV addressing, ASYNCcnt-tracked).
__device__ __forceinline__ void async_cp16B(unsigned ldsByte, const void* gptr) {
    asm volatile("global_load_async_to_lds_b128 %0, %1, off"
                 :: "v"(ldsByte), "v"((unsigned long long)(size_t)gptr)
                 : "memory");
}
__device__ __forceinline__ void wait_async0() {
    asm volatile("s_wait_asynccnt 0x0" ::: "memory");
}

// Tensor Data Mover: DMA a [tile_dim1 x tile_dim0] 2B-element tile (row pitch
// tensor_dim0_stride elements) from global into LDS, +4B pad per 256B row.
// Descriptor per CDNA5 ISA 8.3/8.4; group1 is wave-constant, group0 per call.
__device__ __forceinline__ void tdm_load(const void* gaddr, unsigned ldsByte, v8u g1) {
    unsigned long long a = (unsigned long long)(size_t)gaddr;
    v4u g0;
    g0.x = 1u;                                   // count=1, user mode, no gather
    g0.y = ldsByte;                              // lds_addr
    g0.z = (unsigned)a;                          // global_addr[31:0]
    g0.w = (unsigned)(a >> 32) | (2u << 30);     // global_addr[56:32] | type=2
    asm volatile("tensor_load_to_lds %0, %1" :: "s"(g0), "s"(g1) : "memory");
}

// Read one 16x32 bf16 WMMA A/B fragment from a padded row-major LDS tile.
// ISA 16-bit A layout: lanes 0-15 hold K = {0..7, 16..23}, lanes 16-31 hold K = {8..15, 24..31},
// row M = lane % 16.  Two 16B contiguous runs per lane -> 2x ds_load_b128.
__device__ __forceinline__ v16bf load_frag(const unsigned short* tile, int rowBase, int lane) {
    const int l = lane & 15, g = lane >> 4;
    const unsigned short* p = tile + (rowBase + l) * LDSS + g * 8;
    union { uint4 u[2]; v16bf v; } x;
    x.u[0] = *(const uint4*)(p);
    x.u[1] = *(const uint4*)(p + 16);
    return x.v;
}

// Read a fragment from a TDM-loaded s-major slab [32 s][TRS pitch]: element (k,m) at slab[k*TRS+m].
__device__ __forceinline__ v16bf load_frag_tr(const unsigned short* slab, int colBase, int lane) {
    const int l = lane & 15, g = lane >> 4;
    const int m = colBase + l;
    union { unsigned short h[16]; v16bf v; } x;
#pragma unroll
    for (int j = 0; j < 8; ++j) {
        x.h[j]     = slab[(g * 8 + j) * TRS + m];
        x.h[8 + j] = slab[(16 + g * 8 + j) * TRS + m];
    }
    return x.v;
}

#define WMMA_BF16(A, Bv, C) \
    __builtin_amdgcn_wmma_f32_16x16x32_bf16(false, (A), false, (Bv), (short)0, (C), false, false)

// =====================================================================
// Kernel 1: P = X @ W^T + bias   (X:[M,512] f32, W:[512,512] f32, P bf16)
// 128x128 tile / 256-thread WG (8 waves, wave = 64x32 = 4x2 WMMA tiles)
// =====================================================================
__global__ __launch_bounds__(256) void proj_kernel(
    const float* __restrict__ X, const float* __restrict__ W,
    const float* __restrict__ bias, unsigned short* __restrict__ P)
{
    __shared__ __align__(16) unsigned short At[128 * LDSS];
    __shared__ __align__(16) unsigned short Bt[128 * LDSS];
    const int tid = threadIdx.x;
    const int lane = tid & 31, wid = tid >> 5;
    const int waveM = (wid >> 2) * 64;
    const int waveN = (wid & 3) * 32;
    const int row0 = blockIdx.x * 128;
    const int col0 = blockIdx.y * 128;

    v8f zero = {0.f,0.f,0.f,0.f,0.f,0.f,0.f,0.f};
    v8f acc[4][2];
#pragma unroll
    for (int i = 0; i < 4; ++i)
#pragma unroll
        for (int j = 0; j < 2; ++j) acc[i][j] = zero;

    const int lr = tid >> 1;            // 0..127: tile row
    const int lk = (tid & 1) * 16;      // 0 or 16: K offset
    const float* ga = X + (size_t)(row0 + lr) * Dn + lk;
    const float* gb = W + (size_t)(col0 + lr) * Dn + lk;
    unsigned short* sa = At + lr * LDSS + lk;
    unsigned short* sb = Bt + lr * LDSS + lk;

    for (int k0 = 0; k0 < Dn; k0 += TILE_K) {
        float4 a0 = *(const float4*)(ga);     float4 a1 = *(const float4*)(ga + 4);
        float4 a2 = *(const float4*)(ga + 8); float4 a3 = *(const float4*)(ga + 12);
        float4 b0 = *(const float4*)(gb);     float4 b1 = *(const float4*)(gb + 4);
        float4 b2 = *(const float4*)(gb + 8); float4 b3 = *(const float4*)(gb + 12);
        __builtin_prefetch(ga + TILE_K, 0, 1);
        __builtin_prefetch(gb + TILE_K, 0, 1);
        ga += TILE_K; gb += TILE_K;

        __syncthreads();   // previous iter's LDS reads done
        uint4 pa, pb;
        pa.x = pack2(a0.x, a0.y); pa.y = pack2(a0.z, a0.w);
        pa.z = pack2(a1.x, a1.y); pa.w = pack2(a1.z, a1.w);
        pb.x = pack2(a2.x, a2.y); pb.y = pack2(a2.z, a2.w);
        pb.z = pack2(a3.x, a3.y); pb.w = pack2(a3.z, a3.w);
        *(uint4*)(sa) = pa; *(uint4*)(sa + 8) = pb;
        pa.x = pack2(b0.x, b0.y); pa.y = pack2(b0.z, b0.w);
        pa.z = pack2(b1.x, b1.y); pa.w = pack2(b1.z, b1.w);
        pb.x = pack2(b2.x, b2.y); pb.y = pack2(b2.z, b2.w);
        pb.z = pack2(b3.x, b3.y); pb.w = pack2(b3.z, b3.w);
        *(uint4*)(sb) = pa; *(uint4*)(sb + 8) = pb;
        __syncthreads();

        v16bf af[4], bf[2];
#pragma unroll
        for (int mi = 0; mi < 4; ++mi) af[mi] = load_frag(At, waveM + mi * 16, lane);
#pragma unroll
        for (int ni = 0; ni < 2; ++ni) bf[ni] = load_frag(Bt, waveN + ni * 16, lane);
#pragma unroll
        for (int mi = 0; mi < 4; ++mi)
#pragma unroll
            for (int ni = 0; ni < 2; ++ni)
                acc[mi][ni] = WMMA_BF16(af[mi], bf[ni], acc[mi][ni]);
    }

    // Epilogue: C 16x16 layout: VGPR e -> M = e + 8*(lane/16), N = lane%16
    const int l = lane & 15, g = lane >> 4;
#pragma unroll
    for (int mi = 0; mi < 4; ++mi)
#pragma unroll
        for (int ni = 0; ni < 2; ++ni) {
            const int gn = col0 + waveN + ni * 16 + l;
            const float bv_ = bias[gn];
            const int gm = row0 + waveM + mi * 16 + 8 * g;
#pragma unroll
            for (int e = 0; e < 8; ++e)
                P[(size_t)(gm + e) * Dn + gn] = f2bf(acc[mi][ni][e] + bv_);
        }
}

// =====================================================================
// Kernel 2: numPart[p][b] += Q^T K over S-chunk (deterministic split-K)
// Tiles fetched with the Tensor Data Mover (TENSORcnt), double-buffered:
// wave 0 issues two tensor_load_to_lds per step, all waves consume.
// =====================================================================
__global__ __launch_bounds__(256) void scores_kernel(
    const unsigned short* __restrict__ Qb, const unsigned short* __restrict__ Kb,
    float* __restrict__ numPart)
{
    __shared__ __align__(16) unsigned short AtT[2][32 * TRS];  // [s][col] slabs, 260B pitch
    __shared__ __align__(16) unsigned short BtT[2][32 * TRS];
    const int tid = threadIdx.x;
    const int lane = tid & 31, wid = tid >> 5;
    const int waveM = (wid >> 2) * 64;
    const int waveN = (wid & 3) * 32;
    const int b    = blockIdx.x >> 3;
    const int part = blockIdx.x & 7;
    const int q0 = blockIdx.y * 128;
    const int k0 = blockIdx.z * 128;
    const unsigned short* Qp = Qb + (size_t)b * Sn * Dn + q0;  // column base folded in
    const unsigned short* Kp = Kb + (size_t)b * Sn * Dn + k0;
    const int sBeg = part * (Sn / SPLITK);

    // D# group 1 (wave-constant): mask=0, data_size=2B, pad_enable, pad every
    // 256B (interval code 5) by 1 dword; tensor_dim0=512, tensor_dim1=8192,
    // tile_dim0=128, tile_dim1=32, tensor_dim0_stride=512 elements.
    const v8u g1 = { (1u << 16) | (1u << 20) | (5u << 22),  // data_size|pad_en|pad_interval
                     (unsigned)Dn << 16,                    // tensor_dim0 lo16 -> [63:48]
                     (unsigned)Sn << 16,                    // tensor_dim1 lo16 -> [95:80]
                     128u << 16,                            // tile_dim0 -> [127:112]
                     32u,                                   // tile_dim1 -> [143:128]
                     (unsigned)Dn,                          // tensor_dim0_stride lo32
                     0u, 0u };

    const unsigned ldsA[2] = { (unsigned)(size_t)&AtT[0][0], (unsigned)(size_t)&AtT[1][0] };
    const unsigned ldsB[2] = { (unsigned)(size_t)&BtT[0][0], (unsigned)(size_t)&BtT[1][0] };

    v8f zero = {0.f,0.f,0.f,0.f,0.f,0.f,0.f,0.f};
    v8f acc[4][2];
#pragma unroll
    for (int i = 0; i < 4; ++i)
#pragma unroll
        for (int j = 0; j < 2; ++j) acc[i][j] = zero;

    auto issue = [&](int step) {
        const int buf = step & 1;
        const size_t s = (size_t)(sBeg + step * TILE_K);
        tdm_load(Qp + s * Dn, ldsA[buf], g1);
        tdm_load(Kp + s * Dn, ldsB[buf], g1);
    };

    constexpr int NTS = (Sn / SPLITK) / TILE_K;   // 32 K-steps
    if (wid == 0) issue(0);
    for (int st = 0; st < NTS; ++st) {
        if (wid == 0) __builtin_amdgcn_s_wait_tensorcnt(0);  // slab st landed
        __syncthreads();                                     // visible to all waves
        if (wid == 0 && st + 1 < NTS) issue(st + 1);         // overlap DMA with WMMA
        const unsigned short* As = AtT[st & 1];
        const unsigned short* Bs = BtT[st & 1];

        v16bf af[4], bf[2];
#pragma unroll
        for (int mi = 0; mi < 4; ++mi) af[mi] = load_frag_tr(As, waveM + mi * 16, lane);
#pragma unroll
        for (int ni = 0; ni < 2; ++ni) bf[ni] = load_frag_tr(Bs, waveN + ni * 16, lane);
#pragma unroll
        for (int mi = 0; mi < 4; ++mi)
#pragma unroll
            for (int ni = 0; ni < 2; ++ni)
                acc[mi][ni] = WMMA_BF16(af[mi], bf[ni], acc[mi][ni]);
    }

    float* dst = numPart + ((size_t)part * Bn + b) * Dn * Dn;
    const int l = lane & 15, g = lane >> 4;
#pragma unroll
    for (int mi = 0; mi < 4; ++mi)
#pragma unroll
        for (int ni = 0; ni < 2; ++ni) {
            const int gn = k0 + waveN + ni * 16 + l;
            const int gm = q0 + waveM + mi * 16 + 8 * g;
#pragma unroll
            for (int e = 0; e < 8; ++e)
                dst[(size_t)(gm + e) * Dn + gn] = acc[mi][ni][e];
        }
}

// =====================================================================
// Kernel 3: sum split-K partials, mask, row softmax, emit bf16 attn
// One 256-thread WG per row (b*512 + q), 512 columns.
// =====================================================================
__global__ __launch_bounds__(256) void softmax_kernel(
    const float* __restrict__ numPart, const int* __restrict__ gtok,
    unsigned short* __restrict__ attnb)
{
    __shared__ unsigned int glb[16];   // 512-bit global-token mask
    __shared__ float red[256];
    const int row = blockIdx.x;        // b*Dn + q
    const int qy = row & (Dn - 1);
    const int tid = threadIdx.x;

    if (tid < 16) glb[tid] = 0u;
    __syncthreads();
    if (tid < NGL) { int t = gtok[tid] & (Dn - 1); atomicOr(&glb[t >> 5], 1u << (t & 31)); }
    __syncthreads();
    const bool qg = (glb[qy >> 5] >> (qy & 31)) & 1u;
    const float scale = 0.04419417382415922f;   // 1/sqrt(512)

    float s0 = 0.f, s1 = 0.f;
#pragma unroll
    for (int p = 0; p < SPLITK; ++p) {
        const float* src = numPart + ((size_t)p * Bn * Dn + row) * Dn;
        s0 += src[tid];
        s1 += src[tid + 256];
    }
    int k = tid;
    bool kg = (glb[k >> 5] >> (k & 31)) & 1u;
    float x0 = ((qy < k) && !qg && !kg) ? -__builtin_inff() : s0 * scale;
    k = tid + 256;
    kg = (glb[k >> 5] >> (k & 31)) & 1u;
    float x1 = ((qy < k) && !qg && !kg) ? -__builtin_inff() : s1 * scale;

    red[tid] = fmaxf(x0, x1);
    __syncthreads();
    for (int off = 128; off > 0; off >>= 1) {
        if (tid < off) red[tid] = fmaxf(red[tid], red[tid + off]);
        __syncthreads();
    }
    const float mx = red[0];
    __syncthreads();

    const float e0 = __expf(x0 - mx), e1 = __expf(x1 - mx);
    red[tid] = e0 + e1;
    __syncthreads();
    for (int off = 128; off > 0; off >>= 1) {
        if (tid < off) red[tid] += red[tid + off];
        __syncthreads();
    }
    const float inv = 1.0f / red[0];
    attnb[(size_t)row * Dn + tid]       = f2bf(e0 * inv);
    attnb[(size_t)row * Dn + tid + 256] = f2bf(e1 * inv);
}

// =====================================================================
// Kernel 4: out[b] = attn[b] @ V[b]^T   (M=512 q, N=8192 s, K=512)
// bf16 operands, contiguous K -> async global->LDS DMA (ASYNCcnt),
// double-buffered so tile k+1 DMA overlaps WMMA on tile k.
// =====================================================================
__global__ __launch_bounds__(256) void out_kernel(
    const unsigned short* __restrict__ attn, const unsigned short* __restrict__ Vb,
    float* __restrict__ out)
{
    __shared__ __align__(16) unsigned short At[2][128 * LDSS];
    __shared__ __align__(16) unsigned short Bt[2][128 * LDSS];
    const int tid = threadIdx.x;
    const int lane = tid & 31, wid = tid >> 5;
    const int waveM = (wid >> 2) * 64;
    const int waveN = (wid & 3) * 32;
    const int b = blockIdx.x;
    const int q0 = blockIdx.y * 128;
    const int s0 = blockIdx.z * 128;
    const unsigned short* A = attn + (size_t)b * Dn * Dn;
    const unsigned short* Bm = Vb + (size_t)b * Sn * Dn;

    v8f zero = {0.f,0.f,0.f,0.f,0.f,0.f,0.f,0.f};
    v8f acc[4][2];
#pragma unroll
    for (int i = 0; i < 4; ++i)
#pragma unroll
        for (int j = 0; j < 2; ++j) acc[i][j] = zero;

    const int lr = tid >> 1;            // 0..127: tile row
    const int lk = (tid & 1) * 16;      // 0 or 16: K offset (elements)
    const unsigned short* ga = A + (size_t)(q0 + lr) * Dn + lk;
    const unsigned short* gb = Bm + (size_t)(s0 + lr) * Dn + lk;
    const unsigned ldsA[2] = { (unsigned)(size_t)(&At[0][lr * LDSS + lk]),
                               (unsigned)(size_t)(&At[1][lr * LDSS + lk]) };
    const unsigned ldsB[2] = { (unsigned)(size_t)(&Bt[0][lr * LDSS + lk]),
                               (unsigned)(size_t)(&Bt[1][lr * LDSS + lk]) };

    constexpr int NT = Dn / TILE_K;     // 16 K-tiles

    auto issue = [&](int kt) {
        const int buf = kt & 1;
        const unsigned short* pa = ga + kt * TILE_K;
        const unsigned short* pb = gb + kt * TILE_K;
        async_cp16B(ldsA[buf],      pa);
        async_cp16B(ldsA[buf] + 16, pa + 8);
        async_cp16B(ldsB[buf],      pb);
        async_cp16B(ldsB[buf] + 16, pb + 8);
    };

    issue(0);
    for (int kt = 0; kt < NT; ++kt) {
        wait_async0();                  // this thread's tile-kt DMA done
        __syncthreads();                // ...and everyone else's; prior readers done
        if (kt + 1 < NT) issue(kt + 1); // overlap next DMA with this tile's WMMAs
        const unsigned short* Atb = At[kt & 1];
        const unsigned short* Btb = Bt[kt & 1];

        v16bf af[4], bf[2];
#pragma unroll
        for (int mi = 0; mi < 4; ++mi) af[mi] = load_frag(Atb, waveM + mi * 16, lane);
#pragma unroll
        for (int ni = 0; ni < 2; ++ni) bf[ni] = load_frag(Btb, waveN + ni * 16, lane);
#pragma unroll
        for (int mi = 0; mi < 4; ++mi)
#pragma unroll
            for (int ni = 0; ni < 2; ++ni)
                acc[mi][ni] = WMMA_BF16(af[mi], bf[ni], acc[mi][ni]);
    }

    const int l = lane & 15, g = lane >> 4;
#pragma unroll
    for (int mi = 0; mi < 4; ++mi)
#pragma unroll
        for (int ni = 0; ni < 2; ++ni) {
            const int gn = s0 + waveN + ni * 16 + l;        // sequence position
            const int gm = q0 + waveM + mi * 16 + 8 * g;    // q index
#pragma unroll
            for (int e = 0; e < 8; ++e)
                out[((size_t)b * Dn + gm + e) * Sn + gn] = acc[mi][ni][e];
        }
}

// =====================================================================
extern "C" void kernel_launch(void* const* d_in, const int* in_sizes, int n_in,
                              void* d_out, int out_size, void* d_ws, size_t ws_size,
                              hipStream_t stream) {
    const float* q   = (const float*)d_in[0];
    const float* k   = (const float*)d_in[1];
    const float* v   = (const float*)d_in[2];
    const float* Wq  = (const float*)d_in[3];
    const float* bq  = (const float*)d_in[4];
    const float* Wk  = (const float*)d_in[5];
    const float* bk  = (const float*)d_in[6];
    const float* Wv  = (const float*)d_in[7];
    const float* bv  = (const float*)d_in[8];
    const int* gtok  = (const int*)d_in[9];
    float* out = (float*)d_out;

    // Workspace layout:
    //  Qb/Kb/Vb  : bf16 [B,S,512]     -> 3 * 33,554,432 B
    //  numPart   : f32  [8,B,512,512] -> 33,554,432 B
    //  attnb     : bf16 [B,512,512]   -> 2,097,152 B   (total ~136 MB)
    unsigned short* Qb = (unsigned short*)d_ws;
    unsigned short* Kb = Qb + (size_t)Bn * Sn * Dn;
    unsigned short* Vb = Kb + (size_t)Bn * Sn * Dn;
    float* numPart = (float*)(Vb + (size_t)Bn * Sn * Dn);
    unsigned short* attnb = (unsigned short*)(numPart + (size_t)SPLITK * Bn * Dn * Dn);

    dim3 blk(256);
    dim3 gProj((Bn * Sn) / 128, Dn / 128);          // 256 x 4
    proj_kernel<<<gProj, blk, 0, stream>>>(q, Wq, bq, Qb);
    proj_kernel<<<gProj, blk, 0, stream>>>(k, Wk, bk, Kb);
    proj_kernel<<<gProj, blk, 0, stream>>>(v, Wv, bv, Vb);

    dim3 gScores(Bn * SPLITK, Dn / 128, Dn / 128);  // 32 x 4 x 4 = 512 WGs
    scores_kernel<<<gScores, blk, 0, stream>>>(Qb, Kb, numPart);

    softmax_kernel<<<dim3(Bn * Dn), blk, 0, stream>>>(numPart, gtok, attnb);

    dim3 gOut(Bn, Dn / 128, Sn / 128);              // 4 x 4 x 64 = 1024 WGs
    out_kernel<<<gOut, blk, 0, stream>>>(attnb, Vb, out);
}
